// HeartDisDet_78426102825261
// MI455X (gfx1250) — matrix operand order repaired and checked by reading, not compiled, not run on hardware
//
#include <hip/hip_runtime.h>
#include <hip/hip_bf16.h>
#include <math.h>

typedef __attribute__((ext_vector_type(16))) _Float16 v16h;
typedef __attribute__((ext_vector_type(8)))  _Float16 v8h;
typedef __attribute__((ext_vector_type(8)))  float    v8f;

#define NTHREADS 256
#define ROWS_PER_BLK 64

// LDS: transposed f16 weights (zero-padded K), plus staged activations.
// Total ~156 KB (< 320 KB/WGP).
struct SMem {
  alignas(32) _Float16 W1T[256 * 64];   // [n][k], k 44..63 zero     32 KB
  alignas(32) _Float16 W2T[128 * 256];  // [n][k]                    64 KB
  alignas(32) _Float16 W3T[16 * 128];   // [n][k], n 2..15 zero       4 KB
  alignas(32) _Float16 A0[64 * 64];     // gathered input, K pad      8 KB
  alignas(32) _Float16 A1[64 * 256];    // layer-1 activations       32 KB
  alignas(32) _Float16 A2[64 * 128];    // layer-2 activations       16 KB
};

// Branch-free tanh on the transcendental unit: v_med3 clamp + v_exp + v_rcp.
__device__ __forceinline__ float fast_tanh(float x) {
  x = fminf(fmaxf(x, -15.f), 15.f);
  float e = __builtin_amdgcn_exp2f(x * 2.8853900817779268f);  // exp(2x)
  return (e - 1.f) * __builtin_amdgcn_rcpf(e + 1.f);
}

__device__ __forceinline__ float fast_sigmoid(float x) {
  x = fminf(fmaxf(x, -30.f), 30.f);
  float e = __builtin_amdgcn_exp2f(-x * 1.4426950408889634f);  // exp(-x)
  return __builtin_amdgcn_rcpf(1.f + e);
}

// A-matrix 16x32 f16 fragment (row-major source, stride in halves).
// lanes 0-15: M=lane,   elems 0-7 = K+0..7,  elems 8-15 = K+16..23
// lanes16-31: M=lane-16,elems 0-7 = K+8..15, elems 8-15 = K+24..31
__device__ __forceinline__ v16h load_a_frag(const _Float16* base, int stride,
                                            int rbase, int kbase, int lhalf, int lhi) {
  const _Float16* p = base + (rbase + lhalf) * stride + kbase + lhi * 8;
  v8h lo = *(const v8h*)p;
  v8h hi = *(const v8h*)(p + 16);
  v16h r;
#pragma unroll
  for (int i = 0; i < 8; ++i) { r[i] = lo[i]; r[i + 8] = hi[i]; }
  return r;
}

// B-matrix 32x16 f16 fragment from transposed weights WT[n][k] (stride in halves).
// lanes 0-15: N=lane, K = kbase+0..7,16..23 ; lanes 16-31: N=lane-16, K = kbase+8..15,24..31
__device__ __forceinline__ v16h load_b_frag(const _Float16* base, int stride,
                                            int nbase, int kbase, int lhalf, int lhi) {
  const _Float16* p = base + (nbase + lhalf) * stride + kbase + lhi * 8;
  v8h lo = *(const v8h*)p;
  v8h hi = *(const v8h*)(p + 16);
  v16h r;
#pragma unroll
  for (int i = 0; i < 8; ++i) { r[i] = lo[i]; r[i + 8] = hi[i]; }
  return r;
}

__global__ __launch_bounds__(NTHREADS, 1)
void HeartDisDet_mlp_wmma(const float* __restrict__ conx,
                          const int* __restrict__ cat2,
                          const int* __restrict__ cat3,
                          const int* __restrict__ cat4,
                          const float* __restrict__ emb2,
                          const float* __restrict__ emb3,
                          const float* __restrict__ emb4,
                          const float* __restrict__ W1g, const float* __restrict__ b1g,
                          const float* __restrict__ W2g, const float* __restrict__ b2g,
                          const float* __restrict__ W3g, const float* __restrict__ b3g,
                          float* __restrict__ out, int nrows) {
  __shared__ SMem sm;
  const int tid   = threadIdx.x;
  const int lane  = tid & 31;
  const int wv    = tid >> 5;   // wave id 0..7 (wave32)
  const int lhalf = lane & 15;
  const int lhi   = lane >> 4;  // 0: lanes 0-15, 1: lanes 16-31
  const int ntoff = wv & 1;     // N-tile parity owned by this wave
  const int mt    = wv >> 1;    // M-tile owned by this wave (layers 1-2)

  // ---- Stage weights once per workgroup: f32 -> f16, transposed, K-padded ----
  for (int i = tid; i < 256 * 64; i += NTHREADS) {
    int nn = i >> 6, kk = i & 63;
    sm.W1T[i] = (kk < 44) ? (_Float16)W1g[kk * 256 + nn] : (_Float16)0.f;
  }
  for (int i = tid; i < 128 * 256; i += NTHREADS) {
    int nn = i >> 8, kk = i & 255;
    sm.W2T[i] = (_Float16)W2g[kk * 128 + nn];
  }
  for (int i = tid; i < 16 * 128; i += NTHREADS) {
    int nn = i >> 7, kk = i & 127;
    sm.W3T[i] = (nn < 2) ? (_Float16)W3g[kk * 2 + nn] : (_Float16)0.f;
  }

  // ---- Hoist biases into registers: invariant across all row blocks ----
  float bias1[8], bias2[4];
#pragma unroll
  for (int i = 0; i < 8; ++i) bias1[i] = b1g[(2 * i + ntoff) * 16 + lhalf];
#pragma unroll
  for (int i = 0; i < 4; ++i) bias2[i] = b2g[(2 * i + ntoff) * 16 + lhalf];
  const float bias3 = (lhalf < 2) ? b3g[lhalf] : 0.f;

  __syncthreads();

  const int nblk = (nrows + ROWS_PER_BLK - 1) / ROWS_PER_BLK;
  for (int blk = blockIdx.x; blk < nblk; blk += gridDim.x) {
    const int row0 = blk * ROWS_PER_BLK;

    // ---- Gather: waves 0-1 build A0 (64 rows x 64 K, f16, K 44..63 = 0) ----
    if (tid < ROWS_PER_BLK) {
      const int r = row0 + tid;
      _Float16* dst = &sm.A0[tid * 64];
      if (r < nrows) {
        int f = 0;
#pragma unroll
        for (int i = 0; i < 3; ++i) {
          const int c = cat2[r * 3 + i];
#pragma unroll
          for (int j = 0; j < 4; ++j) dst[f++] = (_Float16)emb2[i * 8 + c * 4 + j];
        }
#pragma unroll
        for (int i = 0; i < 3; ++i) {
          const int c = cat3[r * 3 + i];
#pragma unroll
          for (int j = 0; j < 6; ++j) dst[f++] = (_Float16)emb3[i * 18 + c * 6 + j];
        }
        const int c4 = cat4[r];
#pragma unroll
        for (int j = 0; j < 8; ++j) dst[f++] = (_Float16)emb4[c4 * 8 + j];
#pragma unroll
        for (int j = 0; j < 6; ++j) dst[f++] = (_Float16)conx[r * 6 + j];
#pragma unroll
        for (int j = 44; j < 64; ++j) dst[j] = (_Float16)0.f;
      } else {
#pragma unroll
        for (int j = 0; j < 64; ++j) dst[j] = (_Float16)0.f;
      }
    }
    __syncthreads();

    // ---- Layer 1: (64x64) @ (64x256) -> tanh -> A1 ----
    // Each wave owns M-tile mt; A fragments in registers, 8 N-tiles swept.
    {
      const v16h a0 = load_a_frag(sm.A0, 64, mt * 16, 0, lhalf, lhi);
      const v16h a1 = load_a_frag(sm.A0, 64, mt * 16, 32, lhalf, lhi);
#pragma unroll
      for (int i = 0; i < 8; ++i) {
        const int nt = 2 * i + ntoff;
        v8f c = {};
        v16h bf0 = load_b_frag(sm.W1T, 64, nt * 16, 0, lhalf, lhi);
        c = __builtin_amdgcn_wmma_f32_16x16x32_f16(false, a0, false, bf0, (short)0, c,
                                                   false, false);
        v16h bf1 = load_b_frag(sm.W1T, 64, nt * 16, 32, lhalf, lhi);
        c = __builtin_amdgcn_wmma_f32_16x16x32_f16(false, a1, false, bf1, (short)0, c,
                                                   false, false);
#pragma unroll
        for (int v = 0; v < 8; ++v) {
          const int rr = mt * 16 + v + lhi * 8;
          sm.A1[rr * 256 + nt * 16 + lhalf] = (_Float16)fast_tanh(c[v] + bias1[i]);
        }
      }
    }
    __syncthreads();

    // ---- Layer 2: (64x256) @ (256x128) -> tanh -> A2 ----
    {
      v16h af[8];
#pragma unroll
      for (int kk = 0; kk < 8; ++kk)
        af[kk] = load_a_frag(sm.A1, 256, mt * 16, kk * 32, lhalf, lhi);
#pragma unroll
      for (int i = 0; i < 4; ++i) {
        const int nt = 2 * i + ntoff;
        v8f c = {};
#pragma unroll
        for (int kk = 0; kk < 8; ++kk) {
          v16h bf = load_b_frag(sm.W2T, 256, nt * 16, kk * 32, lhalf, lhi);
          c = __builtin_amdgcn_wmma_f32_16x16x32_f16(false, af[kk], false, bf, (short)0,
                                                     c, false, false);
        }
#pragma unroll
        for (int v = 0; v < 8; ++v) {
          const int rr = mt * 16 + v + lhi * 8;
          sm.A2[rr * 128 + nt * 16 + lhalf] = (_Float16)fast_tanh(c[v] + bias2[i]);
        }
      }
    }
    __syncthreads();

    // ---- Layer 3: (64x128) @ (128x16, cols 2..15 zero) -> sigmoid -> out ----
    if (wv < 4) {
      const int mt3 = wv;
      v8f c = {};
#pragma unroll
      for (int kb = 0; kb < 128; kb += 32) {
        v16h a = load_a_frag(sm.A2, 128, mt3 * 16, kb, lhalf, lhi);
        v16h b = load_b_frag(sm.W3T, 128, 0, kb, lhalf, lhi);
        c = __builtin_amdgcn_wmma_f32_16x16x32_f16(false, a, false, b, (short)0, c,
                                                   false, false);
      }
      if (lhalf < 2) {
#pragma unroll
        for (int v = 0; v < 8; ++v) {
          const int rr = row0 + mt3 * 16 + v + lhi * 8;
          if (rr < nrows)
            out[rr * 2 + lhalf] = fast_sigmoid(c[v] + bias3);
        }
      }
    }
    __syncthreads();  // protect A0/A1/A2 reuse next iteration
  }
}

extern "C" void kernel_launch(void* const* d_in, const int* in_sizes, int n_in,
                              void* d_out, int out_size, void* d_ws, size_t ws_size,
                              hipStream_t stream) {
  (void)n_in; (void)d_ws; (void)ws_size; (void)out_size;
  const float* conx = (const float*)d_in[0];
  const int*   cat2 = (const int*)d_in[1];
  const int*   cat3 = (const int*)d_in[2];
  const int*   cat4 = (const int*)d_in[3];
  const float* emb2 = (const float*)d_in[4];
  const float* emb3 = (const float*)d_in[5];
  const float* emb4 = (const float*)d_in[6];
  const float* W1   = (const float*)d_in[7];
  const float* b1   = (const float*)d_in[8];
  const float* W2   = (const float*)d_in[9];
  const float* b2   = (const float*)d_in[10];
  const float* W3   = (const float*)d_in[11];
  const float* b3   = (const float*)d_in[12];
  float* out = (float*)d_out;

  const int nrows = in_sizes[0] / 6;
  const int nblk  = (nrows + ROWS_PER_BLK - 1) / ROWS_PER_BLK;
  int grid = nblk < 512 ? nblk : 512;  // persistent WGs; weights converted once each

  HeartDisDet_mlp_wmma<<<grid, NTHREADS, 0, stream>>>(
      conx, cat2, cat3, cat4, emb2, emb3, emb4,
      W1, b1, W2, b2, W3, b3, out, nrows);
}
